// TernBinLayer_3315714752879
// MI455X (gfx1250) — 8-bit matrix operand order repaired and checked by reading, compiled, not run on hardware
//
#include <hip/hip_runtime.h>
#include <hip/hip_bf16.h>

typedef __attribute__((ext_vector_type(16))) int   v16i;
typedef __attribute__((ext_vector_type(8)))  float v8f;

#define D_IN  256
#define D_OUT 256

__global__ __launch_bounds__(256) void ternbin_quantW(const float* __restrict__ W,
                                                      unsigned char* __restrict__ wq) {
    const int o    = blockIdx.x * 256 + threadIdx.x;
    const int ct   = o >> 12;
    const int kc   = (o >> 11) & 1;
    const int lane = (o >> 6) & 31;
    const int b    = o & 63;
    const int v    = b >> 2;
    const int byt  = b & 3;
    const int hi   = lane >> 4;
    const int n    = lane & 15;
    const int K    = kc * 128 + (v >> 3) * 64 + ((v >> 1) & 3) * 16 + hi * 8 + (v & 1) * 4 + byt;
    const int col  = ct * 16 + n;
    const float w  = W[K * D_OUT + col];
    wq[o] = (w > 0.5f) ? 0x38 : ((w < -0.5f) ? 0xB8 : 0x00);
}

__global__ __launch_bounds__(256) void ternbin_wmma(const float* __restrict__ x,
                                                    const unsigned char* __restrict__ wq,
                                                    float* __restrict__ out) {
    __shared__ __align__(128) unsigned char ldsA[16384];

    const int t    = threadIdx.x;
    const int lane = t & 31;
    const int wave = t >> 5;
    const long long rowBase = (long long)blockIdx.x * 64;

    {
        const int r  = t >> 2;
        const int q  = t & 3;
        const int rt = r >> 4;
        const int m  = r & 15;
        const float* __restrict__ xr = x + (rowBase + r) * D_IN + q * 64;
#pragma unroll
        for (int i = 0; i < 16; ++i) {
            const float4 f = ((const float4*)xr)[i];
            const unsigned b0 = (f.x >= 0.0f) ? 0x38u : 0xB8u;
            const unsigned b1 = (f.y >= 0.0f) ? 0x38u : 0xB8u;
            const unsigned b2 = (f.z >= 0.0f) ? 0x38u : 0xB8u;
            const unsigned b3 = (f.w >= 0.0f) ? 0x38u : 0xB8u;
            const unsigned packed = b0 | (b1 << 8) | (b2 << 16) | (b3 << 24);
            const int K      = q * 64 + i * 4;
            const int kc     = K >> 7;
            const int Kp     = K & 127;
            const int half64 = Kp >> 6;
            const int K6     = Kp & 63;
            const int hi     = (K6 >> 3) & 1;
            const int j      = K6 >> 4;
            const int lp     = (K6 >> 2) & 1;
            const int v      = half64 * 8 + j * 2 + lp;
            const int dlane  = hi * 16 + m;
            *(unsigned*)&ldsA[rt * 4096 + kc * 2048 + dlane * 64 + v * 4] = packed;
        }
    }
    __syncthreads();

    const int rt = wave & 3;
    const int ch = wave >> 1 >> 1;

    const v16i a0 = *(const v16i*)&ldsA[rt * 4096 + 0 * 2048 + lane * 64];
    const v16i a1 = *(const v16i*)&ldsA[rt * 4096 + 1 * 2048 + lane * 64];

    v8f acc[8];
#pragma unroll
    for (int ct = 0; ct < 8; ++ct) {
        const v8f z = {0.f, 0.f, 0.f, 0.f, 0.f, 0.f, 0.f, 0.f};
        acc[ct] = z;
    }

    const unsigned char* __restrict__ wbase = wq + (ch * 8) * 4096;
#pragma unroll
    for (int ct = 0; ct < 8; ++ct) {
        const v16i b0 = *(const v16i*)&wbase[ct * 4096 + lane * 64];
        acc[ct] = __builtin_amdgcn_wmma_f32_16x16x128_fp8_fp8(
            a0, b0, (short)0, acc[ct], false, false);
    }
#pragma unroll
    for (int ct = 0; ct < 8; ++ct) {
        const v16i b1 = *(const v16i*)&wbase[ct * 4096 + 2048 + lane * 64];
        acc[ct] = __builtin_amdgcn_wmma_f32_16x16x128_fp8_fp8(
            a1, b1, (short)0, acc[ct], false, false);
    }

    const int n = lane & 15;
    const int cg = lane >> 4;
    float* __restrict__ obase =
        out + (rowBase + rt * 16 + cg * 8) * D_OUT + ch * 128 + n;
#pragma unroll
    for (int ct = 0; ct < 8; ++ct) {
#pragma unroll
        for (int r = 0; r < 8; ++r) {
            obase[(long long)r * D_OUT + ct * 16] =
                (acc[ct][r] >= 0.0f) ? 1.0f : -1.0f;
        }
    }
}

extern "C" void kernel_launch(void* const* d_in, const int* in_sizes, int n_in,
                              void* d_out, int out_size, void* d_ws, size_t ws_size,
                              hipStream_t stream) {
    const float* x = (const float*)d_in[0];
    const float* W = (const float*)d_in[1];
    float* out = (float*)d_out;
    unsigned char* wq = (unsigned char*)d_ws;

    const long long rows = (long long)in_sizes[0] / D_IN;
    const int grid = (int)(rows / 64);

    ternbin_quantW<<<256, 256, 0, stream>>>(W, wq);
    ternbin_wmma<<<grid, 256, 0, stream>>>(x, wq, out);
}
